// NeuralSemiLagrangian_23450521436683
// MI455X (gfx1250) — matrix operand order repaired and checked by reading, not compiled, not run on hardware
//
#include <hip/hip_runtime.h>
#include <cstdint>
#include <cstddef>

typedef __attribute__((ext_vector_type(16))) _Float16 v16h;
typedef __attribute__((ext_vector_type(8)))  _Float16 v8h;
typedef __attribute__((ext_vector_type(8)))  float    v8f;

#define BATCH 2
#define CCH   64
#define HH    361
#define WW    720
#define HWSZ  (HH*WW)            // 259920
#define NPIXT (BATCH*HWSZ)       // 519840
#define HP    (HH+4)             // 365
#define WP    (WW+4)             // 724
#define TILES_PER_WG 15
#define PIX_PER_WG   (TILES_PER_WG*16)   // 240
#define NWG          (NPIXT/PIX_PER_WG)  // 2166 (exact)
#define XSTRIDE 136   // f16 elements per pixel row (128 + pad, 272B = conflict-free)
#define PSTRIDE 132   // f32 elements per pixel row for pos

__device__ __forceinline__ void cubic_w(float t, float w[4]) {
    const float A = -0.75f;
    float t0 = t + 1.0f;
    w[0] = ((A*t0 - 5.0f*A)*t0 + 8.0f*A)*t0 - 4.0f*A;
    w[1] = ((A + 2.0f)*t - (A + 3.0f))*t*t + 1.0f;
    float s = 1.0f - t;
    w[2] = ((A + 2.0f)*s - (A + 3.0f))*s*s + 1.0f;
    float t3 = 2.0f - t;
    w[3] = ((A*t3 - 5.0f*A)*t3 + 8.0f*A)*t3 - 4.0f*A;
}

__device__ __forceinline__ float silu_fast(float z) {
    // z * sigmoid(z); v_rcp_f32 (1-ulp) instead of IEEE divide sequence —
    // result is quantized to f16 immediately afterwards anyway.
    return z * __builtin_amdgcn_rcpf(1.0f + __expf(-z));
}

__global__ __launch_bounds__(256)
void nsl_fused_kernel(const float* __restrict__ h0, const float* __restrict__ h1,
                      const float* __restrict__ latg, const float* __restrict__ lonp,
                      const float* __restrict__ w1, const float* __restrict__ b1,
                      const float* __restrict__ w2, const float* __restrict__ b2,
                      float* __restrict__ out)
{
    __shared__ __align__(16) _Float16 xlds[16*XSTRIDE];
    __shared__ __align__(16) _Float16 zlds[16*XSTRIDE];
    __shared__ __align__(16) float    plds[16*PSTRIDE];

    const int t    = threadIdx.x;
    const int wave = t >> 5;        // 0..7 : owns out-channels [16*wave, 16*wave+16)
    const int lane = t & 31;
    const int lo   = lane & 15;
    const int hi   = lane >> 4;

    // ---- A fragments (weights f32 -> f16), held in registers for all tiles ----
    // A 16x32, this code's contiguous k order (B matches): lanes0-15 M=lo K=kb*32+0..15; lanes16-31 K=kb*32+16..31.
    const int arow = wave*16 + lo;
    v16h a1f[4], a2f[4];
    #pragma unroll
    for (int kb = 0; kb < 4; ++kb) {
        const int khalf = kb*32 + hi*16;
        const float* p1 = w1 + arow*128 + khalf;
        const float* p2 = w2 + arow*128 + khalf;
        v16h A1, A2;
        #pragma unroll
        for (int i = 0; i < 8; ++i) {
            A1[i]   = (_Float16)p1[i];
            A1[i+8] = (_Float16)p1[i+8];
            A2[i]   = (_Float16)p2[i];
            A2[i+8] = (_Float16)p2[i+8];
        }
        a1f[kb] = A1; a2f[kb] = A2;
    }
    // bias in C-layout slots: slot r -> out row wave*16 + hi*8 + r
    float bb1[8], bb2[8];
    #pragma unroll
    for (int r = 0; r < 8; ++r) {
        bb1[r] = b1[wave*16 + hi*8 + r];
        bb2[r] = b2[wave*16 + hi*8 + r];
    }

    const float minlon = lonp[0];
    const float maxlon = lonp[WW-1];
    const float minlat = latg[0];
    const float maxlat = latg[(HH-1)*WW];
    const float sxn = 2.0f / (maxlon - minlon);
    const float syn = 2.0f / (maxlat - minlat);

    const int px  = t & 15;     // pixel-in-tile for this thread
    const int grp = t >> 4;     // 0..15
    const int ch0 = grp * 8;    // channel group for x staging

    const int wgBase = blockIdx.x * PIX_PER_WG;

    #pragma unroll 1
    for (int tile = 0; tile < TILES_PER_WG; ++tile) {
        const int tbase = wgBase + tile*16;
        const int p     = tbase + px;
        const int b     = (p >= HWSZ) ? 1 : 0;
        const int phw   = p - b*HWSZ;

        // ---- stage x tile (128ch x 16px) into LDS as f16, layout [px][ch] ----
        {
            const float* src = (ch0 < CCH ? h0 : h1)
                             + (size_t)b*CCH*HWSZ + (size_t)(ch0 & (CCH-1))*HWSZ + phw;
            v8h pk;
            #pragma unroll
            for (int i = 0; i < 8; ++i) pk[i] = (_Float16)src[(size_t)i*HWSZ];
            *(v8h*)&xlds[px*XSTRIDE + ch0] = pk;
        }
        __syncthreads();

        // ---- GEMM1: z = silu(w1 @ x + b1) ----
        // B 32x16, this code's contiguous k order (same as A): lanes0-15 N=lo K=kb*32+0..15; lanes16-31 K=kb*32+16..31.
        v8f acc;
        #pragma unroll
        for (int r = 0; r < 8; ++r) acc[r] = bb1[r];
        #pragma unroll
        for (int kb = 0; kb < 4; ++kb) {
            const _Float16* bp = &xlds[lo*XSTRIDE + kb*32 + hi*16];
            union { v16h v; v8h h2[2]; } u;
            u.h2[0] = *(const v8h*)bp;
            u.h2[1] = *(const v8h*)(bp + 8);
            acc = __builtin_amdgcn_wmma_f32_16x16x32_f16(false, a1f[kb], false, u.v,
                                                         (short)0, acc, false, false);
        }
        #pragma unroll
        for (int r = 0; r < 8; r += 2) {
            float z0 = silu_fast(acc[r]);
            float z1 = silu_fast(acc[r+1]);
            _Float16* zp = &zlds[lo*XSTRIDE + wave*16 + hi*8 + r];
            zp[0] = (_Float16)z0;
            zp[1] = (_Float16)z1;
        }
        __syncthreads();

        // ---- GEMM2: pos = w2 @ z + b2 ----
        v8f acc2;
        #pragma unroll
        for (int r = 0; r < 8; ++r) acc2[r] = bb2[r];
        #pragma unroll
        for (int kb = 0; kb < 4; ++kb) {
            const _Float16* bp = &zlds[lo*XSTRIDE + kb*32 + hi*16];
            union { v16h v; v8h h2[2]; } u;
            u.h2[0] = *(const v8h*)bp;
            u.h2[1] = *(const v8h*)(bp + 8);
            acc2 = __builtin_amdgcn_wmma_f32_16x16x32_f16(false, a2f[kb], false, u.v,
                                                          (short)0, acc2, false, false);
        }
        #pragma unroll
        for (int r = 0; r < 8; ++r)
            plds[lo*PSTRIDE + wave*16 + hi*8 + r] = acc2[r];
        __syncthreads();

        // ---- bicubic sampling with on-the-fly geo-cyclic padding ----
        const float lonv = lonp[phw];
        const float latv = latg[phw];
        #pragma unroll
        for (int it = 0; it < 4; ++it) {
            const int c = grp + 16*it;                       // channel 0..63
            const float posx = plds[px*PSTRIDE + c];         // pos[:,0,c]
            const float posy = plds[px*PSTRIDE + CCH + c];   // pos[:,1,c]

            float gxn = (lonv + posx - minlon)*sxn - 1.0f;
            float gyn = (latv + posy - minlat)*syn - 1.0f;
            float rr  = fmodf(gxn + 1.0f, 2.0f);
            rr += (rr < 0.0f) ? 2.0f : 0.0f;
            float gx = rr - 1.0f;
            const bool left  = (gx <= 0.0f);
            const bool outer = (fabsf(gyn) > 1.0f);
            gx += (outer && left)  ?  1.0f : 0.0f;
            gx += (outer && !left) ? -1.0f : 0.0f;
            float gy = gyn;
            gy = (gy < -1.0f) ? (-2.0f - gy) : gy;
            gy = (gy >  1.0f) ? ( 2.0f - gy) : gy;
            gx *= (float)WW / (float)WP;
            gy *= (float)HH / (float)HP;
            float ix = (gx + 1.0f) * (0.5f * (float)(WP - 1));
            float iy = (gy + 1.0f) * (0.5f * (float)(HP - 1));
            float ix0f = floorf(ix), iy0f = floorf(iy);
            int ix0 = (int)ix0f, iy0 = (int)iy0f;
            float wxv[4], wyv[4];
            cubic_w(ix - ix0f, wxv);
            cubic_w(iy - iy0f, wyv);

            const float* basep = h0 + ((size_t)b*CCH + c)*HWSZ;
            float accv = 0.0f;
            if (ix0 >= 3 && ix0 <= WW-1 && iy0 >= 3 && iy0 <= HH-1) {
                // fully interior: padded(yy,xx) == h0[yy-2][xx-2], contiguous taps
                const float* q = basep + (size_t)(iy0-3)*WW + (ix0-3);
                #pragma unroll
                for (int j = 0; j < 4; ++j) {
                    accv += wyv[j]*(wxv[0]*q[0] + wxv[1]*q[1] + wxv[2]*q[2] + wxv[3]*q[3]);
                    q += WW;
                }
            } else {
                #pragma unroll
                for (int j = 0; j < 4; ++j) {
                    int yy = iy0 - 1 + j;
                    yy = yy < 0 ? 0 : (yy > HP-1 ? HP-1 : yy);
                    int ys, flip;
                    if (yy < 2)          { ys = 1 - yy;        flip = 1; } // pole flip (top)
                    else if (yy >= HH+2) { ys = 2*HH + 1 - yy; flip = 1; } // pole flip (bot)
                    else                 { ys = yy - 2;        flip = 0; }
                    const float* rp = basep + (size_t)ys*WW;
                    float rs = 0.0f;
                    #pragma unroll
                    for (int i = 0; i < 4; ++i) {
                        int xx = ix0 - 1 + i;
                        xx = xx < 0 ? 0 : (xx > WP-1 ? WP-1 : xx);
                        int xc = xx - 2;                       // cyclic lon pad
                        xc += (xc < 0)   ? WW : 0;
                        xc -= (xc >= WW) ? WW : 0;
                        if (flip) xc = (xc >= WW/2) ? (xc - WW/2) : (xc + WW/2); // half-lon roll
                        rs += wxv[i] * rp[xc];
                    }
                    accv += wyv[j] * rs;
                }
            }
            out[((size_t)b*CCH + c)*HWSZ + phw] = accv;
        }
    }
}

extern "C" void kernel_launch(void* const* d_in, const int* in_sizes, int n_in,
                              void* d_out, int out_size, void* d_ws, size_t ws_size,
                              hipStream_t stream) {
    (void)in_sizes; (void)n_in; (void)out_size; (void)d_ws; (void)ws_size;
    const float* h0   = (const float*)d_in[0];
    const float* h1   = (const float*)d_in[1];
    const float* latg = (const float*)d_in[2];
    const float* lng  = (const float*)d_in[3];
    const float* w1   = (const float*)d_in[4];
    const float* b1   = (const float*)d_in[5];
    const float* w2   = (const float*)d_in[6];
    const float* b2   = (const float*)d_in[7];
    float* out = (float*)d_out;
    nsl_fused_kernel<<<dim3(NWG), dim3(256), 0, stream>>>(h0, h1, latg, lng,
                                                          w1, b1, w2, b2, out);
}
